// AlignmentEncoder_10479720202332
// MI455X (gfx1250) — matrix operand order repaired and checked by reading, not compiled, not run on hardware
//
#include <hip/hip_runtime.h>

#define TEMP 0.0005f

typedef __attribute__((ext_vector_type(16))) __bf16 v16bf;
typedef __attribute__((ext_vector_type(8)))  float  v8f;

__device__ __forceinline__ unsigned short f2bf(float f) {
  unsigned u = __float_as_uint(f);
  u += 0x7FFFu + ((u >> 16) & 1u);            // round-to-nearest-even
  return (unsigned short)(u >> 16);
}

union Frag16 { uint4 q[2]; v16bf v; };
union Pack16 { uint4 q[2]; unsigned short s[16]; };

// ---- CDNA5 async global->LDS copy (ASYNCcnt) with safe fallback ------------
#if __has_builtin(__builtin_amdgcn_global_load_async_to_lds_b128)
#define HAVE_ASYNC_LDS 1
typedef int v4i_g __attribute__((vector_size(16)));   // matches builtin's V4i
typedef __attribute__((address_space(1))) v4i_g g1v4i;
typedef __attribute__((address_space(3))) v4i_g l3v4i;
__device__ __forceinline__ void asyncLds128(const void* g, void* l) {
  __builtin_amdgcn_global_load_async_to_lds_b128((g1v4i*)g, (l3v4i*)l, 0, 0);
}
__device__ __forceinline__ void waitAsync0() {
#if __has_builtin(__builtin_amdgcn_s_wait_asynccnt)
  __builtin_amdgcn_s_wait_asynccnt(0);
#else
  asm volatile("s_wait_asynccnt 0x0" ::: "memory");
#endif
}
__device__ __forceinline__ void waitAsync2() {
#if __has_builtin(__builtin_amdgcn_s_wait_asynccnt)
  __builtin_amdgcn_s_wait_asynccnt(2);
#else
  asm volatile("s_wait_asynccnt 0x2" ::: "memory");
#endif
}
#endif

// ---------------------------------------------------------------------------
// Compute phase for one 32-wide K chunk: TAPS A fragments + 4 software-
// pipelined B fragments per tap -> 4*TAPS wmma. GUARD only for K remainder.
// ---------------------------------------------------------------------------
template <int TAPS, bool GUARD>
__device__ __forceinline__ void compute_chunk(
    v8f (&acc)[4], const unsigned short* __restrict__ Wb,
    const unsigned short (*BT)[40],
    int mRow, int M, int K, int c0, int laneHi, int lmod)
{
  constexpr int pad = TAPS >> 1;
  #pragma unroll
  for (int d = 0; d < TAPS; ++d) {
    Frag16 a;
    {
      const unsigned short* wr = Wb + ((size_t)d * M + mRow) * K;
      const int cA = c0 + laneHi * 8;
      if (GUARD) {
        const bool g0 = (cA + 8  <= K);
        const bool g1 = (cA + 24 <= K);
        const uint4 t0 = *(const uint4*)(wr + (g0 ? cA        : K - 8));
        const uint4 t1 = *(const uint4*)(wr + (g1 ? (cA + 16) : K - 8));
        const uint4 z  = make_uint4(0u, 0u, 0u, 0u);
        a.q[0] = g0 ? t0 : z;                   // data-select, never ptr-select
        a.q[1] = g1 ? t1 : z;
      } else {
        a.q[0] = *(const uint4*)(wr + cA);
        a.q[1] = *(const uint4*)(wr + cA + 16);
      }
    }
    const int rowOff = 1 + d - pad;             // tap shift into 66-row tile
    Frag16 bfr[2];
    {
      const unsigned short* r0 = &BT[rowOff + lmod][laneHi * 8];
      bfr[0].q[0] = *(const uint4*)(r0);
      bfr[0].q[1] = *(const uint4*)(r0 + 16);
    }
    #pragma unroll
    for (int i = 0; i < 4; ++i) {
      if (i < 3) {                              // preload next B fragment
        const unsigned short* rn = &BT[rowOff + (i + 1) * 16 + lmod][laneHi * 8];
        bfr[(i + 1) & 1].q[0] = *(const uint4*)(rn);
        bfr[(i + 1) & 1].q[1] = *(const uint4*)(rn + 16);
      }
      acc[i] = __builtin_amdgcn_wmma_f32_16x16x32_bf16(
          false, a.v, false, bfr[i & 1].v, (short)0, acc[i], false, false);
    }
  }
}

// ---------------------------------------------------------------------------
// Generic bf16 WMMA GEMM with fused TAPS-tap conv, bias, ReLU.
//   Y[b][m][n] = act( bias[m] + sum_d sum_k W[d][m][k] * X[b][k][n + d - pad] )
// W: [TAPS][M][K] bf16 (wStride = per-batch stride, 0 if shared)
// X: [K][N] bf16 per batch; out-of-range t (conv padding) reads zero.
// Outputs (each optional): f32 [M][N], bf16 [M][N], bf16 transposed [N][M].
// 128 threads (4 waves); 64x64 macro tile; wave = 16x64 strip, 4 accumulators.
// Interior tiles: double-buffered GLOBAL_LOAD_ASYNC_TO_LDS_B128 raw-tile ring
// (XT) + LDS transpose-scatter into padded BT (conflict-free ds_load_b128).
// ---------------------------------------------------------------------------
template <int TAPS>
__global__ __launch_bounds__(128) void gemm_bf16_wmma(
    const unsigned short* __restrict__ W, long long wStride,
    const unsigned short* __restrict__ X, long long xStride,
    const float* __restrict__ bias,
    float* __restrict__ outF, long long ofStride,
    unsigned short* __restrict__ outB, long long obStride,
    unsigned short* __restrict__ outT, long long otStride,
    int M, int N, int K, int relu)
{
  __shared__ __align__(16) unsigned short BT[66][40];  // transposed, padded
#ifdef HAVE_ASYNC_LDS
  __shared__ __align__(16) unsigned short XT[2][32][64]; // raw async ring
#endif

  const int b      = blockIdx.z;
  const unsigned short* Wb = W + (size_t)b * wStride;
  const unsigned short* Xb = X + (size_t)b * xStride;
  const int tid    = threadIdx.x;
  const int lane   = tid & 31;
  const int laneHi = lane >> 4;
  const int lmod   = lane & 15;
  const int nBase  = blockIdx.x * 64;
  const int mBase  = blockIdx.y * 64 + (tid >> 5) * 16;
  int mRow = mBase + lmod; if (mRow >= M) mRow = M - 1;  // clamp, store-guarded
  const int nkFull = K >> 5;
  const int sk  = tid >> 2;         // staging: K row 0..31
  const int sq  = tid & 3;          // staging: column quarter 0..3

  const bool interior = (nBase >= 1) && (nBase + 64 <= N);   // uniform
  const bool hiEdge   = (nBase + 64) < N;

  v8f acc[4] = {};

#ifdef HAVE_ASYNC_LDS
  if (interior) {
    // async double-buffered main loop: each thread copies 2 b128 segments of
    // its own wave's rows (per-wave ASYNCcnt ordering is sufficient).
    auto issue = [&](int buf, int c0) {
      const unsigned short* g = Xb + (size_t)(c0 + sk) * N + nBase + sq * 8;
      asyncLds128(g,      &XT[buf][sk][sq * 8]);
      asyncLds128(g + 32, &XT[buf][sk][sq * 8 + 32]);
    };
    if (nkFull > 0) issue(0, 0);
    int buf = 0;
    for (int kc = 0; kc < nkFull; ++kc) {
      const int c0 = kc << 5;
      const bool more = (kc + 1) < nkFull;
      if (more) issue(buf ^ 1, c0 + 32);        // prefetch next chunk
      if (more) waitAsync2(); else waitAsync0();
      __syncthreads();                          // previous BT consumed
      {                                         // LDS->LDS transpose scatter
        Pack16 p;
        const unsigned short* xr = &XT[buf][sk][sq * 16];
        p.q[0] = *(const uint4*)(xr);
        p.q[1] = *(const uint4*)(xr + 8);
        #pragma unroll
        for (int e = 0; e < 16; ++e) BT[1 + sq * 16 + e][sk] = p.s[e];
        const int kk = c0 + sk;
        if (sq == 0) BT[0][sk]  = Xb[(size_t)kk * N + nBase - 1];
        if (sq == 1) BT[65][sk] = hiEdge ? Xb[(size_t)kk * N + nBase + 64]
                                         : (unsigned short)0;
      }
      __syncthreads();
      compute_chunk<TAPS, false>(acc, Wb, BT, mRow, M, K, c0, laneHi, lmod);
      buf ^= 1;
    }
  } else
#endif
  {
    for (int kc = 0; kc < nkFull; ++kc) {
      const int c0 = kc << 5;
      const int kk = c0 + sk;
      __syncthreads();
      if (interior) {
        // vectorized staging: 2 aligned b128 loads + transpose scatter
        const unsigned short* xr = Xb + (size_t)kk * N + nBase + sq * 16;
        Pack16 p;
        p.q[0] = *(const uint4*)(xr);
        p.q[1] = *(const uint4*)(xr + 8);
        #pragma unroll
        for (int e = 0; e < 16; ++e) BT[1 + sq * 16 + e][sk] = p.s[e];
        if (sq == 0) BT[0][sk]  = Xb[(size_t)kk * N + nBase - 1];
        if (sq == 1) BT[65][sk] = hiEdge ? Xb[(size_t)kk * N + nBase + 64]
                                         : (unsigned short)0;
      } else {
        // boundary n-tile: branchless guarded scalar staging
        const unsigned short* xr = Xb + (size_t)kk * N;
        const int jBase = sq * 17;
        #pragma unroll
        for (int jj = 0; jj < 17; ++jj) {
          const int j = jBase + jj;
          const int t = nBase - 1 + j;
          const bool ok = (t >= 0) & (t < N) & (j < 66);
          const unsigned short v = xr[ok ? t : 0];
          if (j < 66) BT[j][sk] = ok ? v : (unsigned short)0;
        }
      }
      if (kk + 32 < K) {                        // next-chunk cache hint
        const int pt = nBase - 1 + sq * 17;
        if (pt >= 0 && pt < N)
          __builtin_prefetch((const void*)(Xb + (size_t)(kk + 32) * N + pt), 0, 1);
      }
      __syncthreads();
      compute_chunk<TAPS, false>(acc, Wb, BT, mRow, M, K, c0, laneHi, lmod);
    }
  }

  if (K & 31) {                                 // one guarded remainder chunk
    const int c0 = nkFull << 5;
    __syncthreads();
    {
      const int  kk = c0 + sk;
      const bool kv = kk < K;
      const unsigned short* xr = Xb + (size_t)(kv ? kk : 0) * N;
      const int jBase = sq * 17;
      #pragma unroll
      for (int jj = 0; jj < 17; ++jj) {
        const int j = jBase + jj;
        const int t = nBase - 1 + j;
        const bool ok = kv & (t >= 0) & (t < N) & (j < 66);
        const unsigned short v = xr[ok ? t : 0];
        if (j < 66) BT[j][sk] = ok ? v : (unsigned short)0;
      }
    }
    __syncthreads();
    compute_chunk<TAPS, true>(acc, Wb, BT, mRow, M, K, c0, laneHi, lmod);
  }

  // Epilogue. C/D layout: VGPR r -> M = r + 8*laneHi, N = lane%16.
  float*          outFb = outF ? outF + (size_t)b * ofStride : nullptr;
  unsigned short* outBb = outB ? outB + (size_t)b * obStride : nullptr;
  unsigned short* outTb = outT ? outT + (size_t)b * otStride : nullptr;
  const int mC = mBase + laneHi * 8;
  const int nC = nBase + lmod;
  float bv[8];
  #pragma unroll
  for (int r = 0; r < 8; ++r) {
    int mb = mC + r; if (mb >= M) mb = M - 1;   // clamped (store-guarded)
    bv[r] = bias ? bias[mb] : 0.0f;
  }
  #pragma unroll
  for (int i = 0; i < 4; ++i) {
    const int n = nC + i * 16;
    if (n >= N) continue;
    #pragma unroll
    for (int r = 0; r < 8; ++r) {
      const int m = mC + r;
      if (m >= M) continue;
      float v = acc[i][r] + bv[r];
      if (relu) v = fmaxf(v, 0.0f);
      if (outFb) outFb[(size_t)m * N + n] = v;
      if (outBb) outBb[(size_t)m * N + n] = f2bf(v);
      if (outTb) outTb[(size_t)n * M + m] = f2bf(v);
    }
  }
}

// ---------------------------------------------------------------------------
__global__ void cast_f32_bf16_k(const float* __restrict__ s,
                                unsigned short* __restrict__ d, long long n) {
  long long i = (long long)blockIdx.x * blockDim.x + threadIdx.x;
  if (i < n) d[i] = f2bf(s[i]);
}

// src [Cout][Cin][Kt] f32 -> dst [Kt][Cout][Cin] bf16
__global__ void conv_w_relayout_k(const float* __restrict__ s,
                                  unsigned short* __restrict__ d,
                                  int Cout, int Cin, int Kt) {
  long long n = (long long)Cout * Cin * Kt;
  long long i = (long long)blockIdx.x * blockDim.x + threadIdx.x;
  if (i >= n) return;
  int dd = (int)(i % Kt);
  long long oi = i / Kt;
  int ci = (int)(oi % Cin);
  int co = (int)(oi / Cin);
  d[((size_t)dd * Cout + co) * Cin + ci] = f2bf(s[i]);
}

// out[b][t] = sum_c enc[b][c][t]^2   (enc is f32 [B][C][T])
__global__ void rowsq_k(const float* __restrict__ enc, float* __restrict__ out,
                        int B, int C, int T) {
  long long i = (long long)blockIdx.x * blockDim.x + threadIdx.x;
  if (i >= (long long)B * T) return;
  int b = (int)(i / T), t = (int)(i % T);
  const float* p = enc + (size_t)b * C * T + t;
  float s = 0.f;
  for (int c = 0; c < C; ++c) { float v = p[(size_t)c * T]; s += v * v; }
  out[i] = s;
}

// ---------------------------------------------------------------------------
// wave32 shuffle reductions + 8-partial cross-wave combine
// ---------------------------------------------------------------------------
__device__ __forceinline__ float wredMax(float v) {
  #pragma unroll
  for (int o = 16; o > 0; o >>= 1) v = fmaxf(v, __shfl_xor(v, o, 32));
  return v;
}
__device__ __forceinline__ float wredSum(float v) {
  #pragma unroll
  for (int o = 16; o > 0; o >>= 1) v += __shfl_xor(v, o, 32);
  return v;
}
__device__ __forceinline__ float blockMax(float v, float* part, int wid, int lane) {
  float w = wredMax(v);
  __syncthreads();
  if (lane == 0) part[wid] = w;
  __syncthreads();
  float r = part[0];
  #pragma unroll
  for (int i = 1; i < 8; ++i) r = fmaxf(r, part[i]);
  return r;
}
__device__ __forceinline__ float blockSum(float v, float* part, int wid, int lane) {
  float w = wredSum(v);
  __syncthreads();
  if (lane == 0) part[wid] = w;
  __syncthreads();
  float r = part[0];
  #pragma unroll
  for (int i = 1; i < 8; ++i) r += part[i];
  return r;
}

// ---------------------------------------------------------------------------
// Per-row (b, t_de): dist -> log_softmax + log(prior) -> attn_logprob (in
// place over the qk scratch), then masked softmax -> attn.
// Assumes Ten == 512, blockDim == 256 (2 columns/thread, 8 waves).
// ---------------------------------------------------------------------------
__global__ __launch_bounds__(256) void attn_row_k(
    const float* __restrict__ qsq, const float* __restrict__ ksq,
    const float* __restrict__ prior, const unsigned char* __restrict__ mask,
    float* __restrict__ attn, float* __restrict__ logprob, int Tde, int Ten) {
  __shared__ float part[8];
  const int row  = blockIdx.x;          // b*Tde + t
  const int b    = row / Tde;
  const int j    = threadIdx.x;
  const int wid  = j >> 5, lane = j & 31;
  const int n0   = j, n1 = j + 256;
  float* lpRow   = logprob + (size_t)row * Ten;   // holds qk on entry
  const float qs = qsq[row];
  const float* ks = ksq + (size_t)b * Ten;
  const float l0 = -TEMP * (qs + ks[n0] - 2.f * lpRow[n0]);
  const float l1 = -TEMP * (qs + ks[n1] - 2.f * lpRow[n1]);

  const float mx  = blockMax(fmaxf(l0, l1), part, wid, lane);
  const float sm  = blockSum(expf(l0 - mx) + expf(l1 - mx), part, wid, lane);
  const float lse = mx + logf(sm);

  const float* pr = prior + (size_t)row * Ten;
  const float lp0 = l0 - lse + logf(pr[n0] + 1e-8f);
  const float lp1 = l1 - lse + logf(pr[n1] + 1e-8f);
  lpRow[n0] = lp0; lpRow[n1] = lp1;     // all qk reads completed before here

  const unsigned char* mk = mask + (size_t)b * Ten;
  const float ninf = -__builtin_inff();
  const float ml0 = mk[n0] ? lp0 : ninf;
  const float ml1 = mk[n1] ? lp1 : ninf;
  const float mx2 = blockMax(fmaxf(ml0, ml1), part, wid, lane);
  const float e0 = expf(ml0 - mx2), e1 = expf(ml1 - mx2);
  const float inv = 1.0f / blockSum(e0 + e1, part, wid, lane);
  float* aRow = attn + (size_t)row * Ten;
  aRow[n0] = e0 * inv; aRow[n1] = e1 * inv;
}

// ---------------------------------------------------------------------------
extern "C" void kernel_launch(void* const* d_in, const int* in_sizes, int n_in,
                              void* d_out, int out_size, void* d_ws, size_t ws_size,
                              hipStream_t stream) {
  (void)in_sizes; (void)n_in; (void)out_size; (void)ws_size;
  const int B = 16, Cq = 80, Ck = 512, Ca = 80, Tde = 2000, Ten = 512;
  const int H1 = 2 * Ck;   // 1024
  const int H2 = 2 * Cq;   // 160

  const float* queries = (const float*)d_in[0];
  const float* keys    = (const float*)d_in[1];
  const unsigned char* mask = (const unsigned char*)d_in[2];
  const float* prior   = (const float*)d_in[3];
  const float* kp_w1 = (const float*)d_in[4];
  const float* kp_b1 = (const float*)d_in[5];
  const float* kp_w2 = (const float*)d_in[6];
  const float* kp_b2 = (const float*)d_in[7];
  const float* qp_w1 = (const float*)d_in[8];
  const float* qp_b1 = (const float*)d_in[9];
  const float* qp_w2 = (const float*)d_in[10];
  const float* qp_b2 = (const float*)d_in[11];
  const float* qp_w3 = (const float*)d_in[12];
  const float* qp_b3 = (const float*)d_in[13];

  char* ws = (char*)d_ws;
  size_t off = 0;
  auto alloc = [&](size_t bytes) -> char* {
    char* p = ws + off; off = (off + bytes + 255) & ~(size_t)255; return p;
  };
  unsigned short* keysBf    = (unsigned short*)alloc((size_t)B*Ck*Ten*2);
  unsigned short* hidden1   = (unsigned short*)alloc((size_t)B*H1*Ten*2);
  float*          kencF     = (float*)         alloc((size_t)B*Ca*Ten*4);
  unsigned short* kencB     = (unsigned short*)alloc((size_t)B*Ca*Ten*2);
  unsigned short* queriesBf = (unsigned short*)alloc((size_t)B*Cq*Tde*2);
  unsigned short* q1        = (unsigned short*)alloc((size_t)B*H2*Tde*2);
  unsigned short* q2        = (unsigned short*)alloc((size_t)B*Cq*Tde*2);
  float*          qencF     = (float*)         alloc((size_t)B*Ca*Tde*4);
  unsigned short* qeT       = (unsigned short*)alloc((size_t)B*Tde*Ca*2);  // [t][c]
  float*          qsq       = (float*)         alloc((size_t)B*Tde*4);
  float*          ksq       = (float*)         alloc((size_t)B*Ten*4);
  unsigned short* w1t       = (unsigned short*)alloc((size_t)3*H1*Ck*2);
  unsigned short* w2b       = (unsigned short*)alloc((size_t)Ca*H1*2);
  unsigned short* qw1t      = (unsigned short*)alloc((size_t)3*H2*Cq*2);
  unsigned short* qw2b      = (unsigned short*)alloc((size_t)Cq*H2*2);
  unsigned short* qw3b      = (unsigned short*)alloc((size_t)Ca*Cq*2);

  float* attnOut = (float*)d_out;
  float* lpOut   = attnOut + (size_t)B * Tde * Ten;   // qk scratch, then logprob

  auto cg = [](long long n) { return dim3((unsigned)((n + 255) / 256)); };

  // --- casts / weight relayout ---
  cast_f32_bf16_k<<<cg((long long)B*Ck*Ten), 256, 0, stream>>>(keys, keysBf, (long long)B*Ck*Ten);
  cast_f32_bf16_k<<<cg((long long)B*Cq*Tde), 256, 0, stream>>>(queries, queriesBf, (long long)B*Cq*Tde);
  conv_w_relayout_k<<<cg((long long)H1*Ck*3), 256, 0, stream>>>(kp_w1, w1t, H1, Ck, 3);
  cast_f32_bf16_k<<<cg((long long)Ca*H1), 256, 0, stream>>>(kp_w2, w2b, (long long)Ca*H1);
  conv_w_relayout_k<<<cg((long long)H2*Cq*3), 256, 0, stream>>>(qp_w1, qw1t, H2, Cq, 3);
  cast_f32_bf16_k<<<cg((long long)Cq*H2), 256, 0, stream>>>(qp_w2, qw2b, (long long)Cq*H2);
  cast_f32_bf16_k<<<cg((long long)Ca*Cq), 256, 0, stream>>>(qp_w3, qw3b, (long long)Ca*Cq);

  auto gemm = [&](int taps, const unsigned short* W, long long wS,
                  const unsigned short* X, long long xS, const float* bias,
                  float* oF, long long oFS, unsigned short* oB, long long oBS,
                  unsigned short* oT, long long oTS,
                  int M, int N, int K, int relu) {
    dim3 g((unsigned)((N + 63) / 64), (unsigned)((M + 63) / 64), (unsigned)B);
    if (taps == 3)
      gemm_bf16_wmma<3><<<g, 128, 0, stream>>>(W, wS, X, xS, bias,
                                               oF, oFS, oB, oBS, oT, oTS, M, N, K, relu);
    else
      gemm_bf16_wmma<1><<<g, 128, 0, stream>>>(W, wS, X, xS, bias,
                                               oF, oFS, oB, oBS, oT, oTS, M, N, K, relu);
  };

  // key_proj: conv(512->1024,k3)+ReLU -> conv(1024->80,k1)
  gemm(3, w1t, 0, keysBf, (long long)Ck*Ten, kp_b1,
       nullptr, 0, hidden1, (long long)H1*Ten, nullptr, 0, H1, Ten, Ck, 1);
  gemm(1, w2b, 0, hidden1, (long long)H1*Ten, kp_b2,
       kencF, (long long)Ca*Ten, kencB, (long long)Ca*Ten, nullptr, 0, Ca, Ten, H1, 0);
  // query_proj: conv(80->160,k3)+ReLU -> conv(160->80,k1)+ReLU -> conv(80->80,k1)
  gemm(3, qw1t, 0, queriesBf, (long long)Cq*Tde, qp_b1,
       nullptr, 0, q1, (long long)H2*Tde, nullptr, 0, H2, Tde, Cq, 1);
  gemm(1, qw2b, 0, q1, (long long)H2*Tde, qp_b2,
       nullptr, 0, q2, (long long)Cq*Tde, nullptr, 0, Cq, Tde, H2, 1);
  gemm(1, qw3b, 0, q2, (long long)Cq*Tde, qp_b3,
       qencF, (long long)Ca*Tde, nullptr, 0, qeT, (long long)Tde*Ca, Ca, Tde, Cq, 0);
  // qk = Q^T x K  -> into the attn_logprob half of d_out (scratch)
  gemm(1, qeT, (long long)Tde*Ca, kencB, (long long)Ca*Ten, nullptr,
       lpOut, (long long)Tde*Ten, nullptr, 0, nullptr, 0, Tde, Ten, Ca, 0);

  // squared norms (f32 precision)
  rowsq_k<<<cg((long long)B*Ten), 256, 0, stream>>>(kencF, ksq, B, Ca, Ten);
  rowsq_k<<<cg((long long)B*Tde), 256, 0, stream>>>(qencF, qsq, B, Ca, Tde);

  // fused dist -> log_softmax+prior -> masked softmax
  attn_row_k<<<dim3((unsigned)(B*Tde)), 256, 0, stream>>>(
      qsq, ksq, prior, mask, attnOut, lpOut, Tde, Ten);
}
